// DiTMoEBlock_59966333386970
// MI455X (gfx1250) — matrix operand order repaired and checked by reading, not compiled, not run on hardware
//
#include <hip/hip_runtime.h>
#include <cstdint>
#include <cstddef>

// ---------------------------------------------------------------------------
// DiT-MoE block for MI455X (gfx1250, wave32, WMMA).
// Compute-bound (~95 GFLOP after top-2 MoE gather vs ~120MB bf16 weights at
// 23.3 TB/s) -> all GEMMs on v_wmma_f32_16x16x32_bf16, f32 accumulation.
// Fragment feeds are ds_load_b128-based; MoE weight tiles are staged with the
// Tensor Data Mover, double-buffered (issue tile k+1, s_wait_tensorcnt 2,
// compute tile k) so the DMA overlaps the matrix pipe.
// ---------------------------------------------------------------------------

typedef __bf16 bf16;
typedef __attribute__((ext_vector_type(16))) __bf16 v16bf;
typedef __attribute__((ext_vector_type(8)))  __bf16 v8bf;
typedef __attribute__((ext_vector_type(8)))  float  v8f;
typedef __attribute__((ext_vector_type(4)))  unsigned int u32x4;
typedef __attribute__((ext_vector_type(8)))  int i32x8;
typedef __attribute__((ext_vector_type(4)))  int i32x4;

#define DEV __device__ __forceinline__

#if defined(__has_builtin)
#if __has_builtin(__builtin_amdgcn_tensor_load_to_lds) && \
    __has_builtin(__builtin_amdgcn_s_wait_tensorcnt)
#define HAVE_TDM 1
#endif
#endif
#ifndef HAVE_TDM
#define HAVE_TDM 0
#endif

static constexpr int Bc = 2, Nc = 1024, Dc = 1024, Hc = 16, HDc = 64;
static constexpr int Ec = 8, Kt = 2, Fc = 2048;
static constexpr int Tc = Bc * Nc;            // 2048 tokens
static constexpr int SIXD = 6 * Dc;           // 6144
static constexpr int MAXSLOTS = Tc * Kt + Ec * 64;  // 4608

static constexpr int LDA = 40;   // A-tile LDS stride (halfs); 80B, 16B aligned
static constexpr int LDB = 40;   // B-tile (n-major) LDS stride
static constexpr int LDT = 72;   // 64-wide attention tiles; 144B, 16B aligned
static constexpr int LDS_S = 68; // f32 score tile stride; 272B, 16B aligned

DEV v8f zero8() {
  v8f z;
#pragma unroll
  for (int i = 0; i < 8; ++i) z[i] = 0.f;
  return z;
}

DEV v8f wmma_bf16(v16bf a, v16bf b, v8f c) {
  return __builtin_amdgcn_wmma_f32_16x16x32_bf16(false, a, false, b, (short)0, c,
                                                 false, false);
}

DEV v16bf join8(v8bf lo, v8bf hi) {
  return __builtin_shufflevector(lo, hi, 0, 1, 2, 3, 4, 5, 6, 7,
                                 8, 9, 10, 11, 12, 13, 14, 15);
}

// A fragment (16x32 bf16) from row-major LDS tile.
// lane (m=lane&15, lg=lane>>4): halves h -> k = (h&7)+8*lg+(h&8?16:0)
// == two contiguous 8-half (16B) runs at k = 8*lg and k = 16+8*lg.
DEV v16bf fragA(const bf16* s, int ld, int rowBase, int kBase, int lane) {
  const int lg = lane >> 4, m = lane & 15;
  const bf16* p = s + (rowBase + m) * ld + kBase;
  const v8bf lo = *(const v8bf*)(p + 8 * lg);
  const v8bf hi = *(const v8bf*)(p + 16 + 8 * lg);
  return join8(lo, hi);
}

// B fragment (32x16 bf16) from n-major LDS tile (Bs[n][k]).
// lane (n=lane&15, lg=lane>>4): halves h -> k = (h&7)+8*lg+(h&8?16:0)
// == two contiguous 8-half (16B) runs at k = 8*lg and k = 16+8*lg.
DEV v16bf fragB(const bf16* s, int ld, int nBase, int kBase, int lane) {
  const int lg = lane >> 4, n = lane & 15;
  const bf16* p = s + (nBase + n) * ld + kBase;
  const v8bf lo = *(const v8bf*)(p + 8 * lg);
  const v8bf hi = *(const v8bf*)(p + 16 + 8 * lg);
  return join8(lo, hi);
}

union U128 { uint4 q; v8bf v; bf16 h[8]; };

// ---------------------------------------------------------------------------
// TDM: DMA one 2D tile (rows x 32 halfs, row stride = strideHalfs) from
// global to LDS, padded +8 halfs per row to match the LDS stride of 40.
// D# layout per CDNA5 ISA 8.3/8.4: group0 = {flags, lds_addr, global_addr,
// type=2}; group1 = {data_size=2B, pad_enable, pad_interval=16dw,
// pad_amount=4dw, tensor dims, tile dims, dim0 stride}.
// Issue from a single wave (TDM ignores EXEC); wait with s_wait_tensorcnt.
// ---------------------------------------------------------------------------
DEV void tdm_load_2d(const void* gsrc, void* ldsDst, int rows, int strideHalfs) {
#if HAVE_TDM
  const unsigned long long ga = (unsigned long long)(uintptr_t)gsrc;
  u32x4 g0;
  g0[0] = 1u;                                    // count=1, user mode
  g0[1] = (unsigned int)(uintptr_t)ldsDst;       // LDS byte address (addr[31:0])
  g0[2] = (unsigned int)ga;                      // global_addr[31:0]
  g0[3] = (unsigned int)((ga >> 32) & 0x1FFFFFFu) | (2u << 30);  // type=2
  const unsigned int td0 = (unsigned int)strideHalfs;  // tensor dim0 (elems)
  const unsigned int td1 = 1u << 20;                   // plenty of rows
  i32x8 g1;
  g1[0] = (int)((1u << 16) |      // data_size = 2 bytes
                (1u << 20) |      // pad_enable
                (3u << 22) |      // pad_interval: 16 dwords (one 32-half row)
                (3u << 25));      // pad_amount: 4 dwords (16B -> stride 40)
  g1[1] = (int)((td0 & 0xFFFFu) << 16);                 // tensor_dim0[15:0]
  g1[2] = (int)((td0 >> 16) | ((td1 & 0xFFFFu) << 16)); // dim0 hi | dim1 lo
  g1[3] = (int)((td1 >> 16) | (32u << 16));             // dim1 hi | tile_dim0=32
  g1[4] = rows;                                         // tile_dim1
  g1[5] = (int)td0;                                     // dim0 stride[31:0]
  g1[6] = 0;
  g1[7] = 0;
  const i32x4 z4 = {0, 0, 0, 0};
#if __clang_major__ >= 23
  const i32x8 z8 = {0, 0, 0, 0, 0, 0, 0, 0};
  __builtin_amdgcn_tensor_load_to_lds(g0, g1, z4, z4, z8, 0);
#else
  __builtin_amdgcn_tensor_load_to_lds(g0, g1, z4, z4, 0);
#endif
#else
  (void)gsrc; (void)ldsDst; (void)rows; (void)strideHalfs;
#endif
}

DEV void tdm_wait0() {
#if HAVE_TDM
  __builtin_amdgcn_s_wait_tensorcnt(0);
#endif
}
DEV void tdm_wait2() {
#if HAVE_TDM
  __builtin_amdgcn_s_wait_tensorcnt(2);
#endif
}

// ---------------------------------------------------------------------------
// Utility kernels
// ---------------------------------------------------------------------------
__global__ void k_cvt(const float* __restrict__ s, bf16* __restrict__ d, size_t n) {
  size_t i = (size_t)blockIdx.x * blockDim.x + threadIdx.x;
  const size_t stride = (size_t)gridDim.x * blockDim.x;
  for (; i < n; i += stride) d[i] = (bf16)s[i];
}

__global__ void k_zero_bf16(bf16* p, int n) {
  int i = blockIdx.x * blockDim.x + threadIdx.x;
  if (i < n) p[i] = (bf16)0.f;
}

__global__ void k_zero_i32(int* p, int n) {
  int i = blockIdx.x * blockDim.x + threadIdx.x;
  if (i < n) p[i] = 0;
}

// ---------------------------------------------------------------------------
// AdaLN: emb[b, :] = silu(t[b,:]) @ ada_w + ada_b       (tiny: 2 x 6144)
// ---------------------------------------------------------------------------
__global__ __launch_bounds__(256) void k_adaln(const float* __restrict__ t,
                                               const float* __restrict__ W,
                                               const float* __restrict__ bias,
                                               float* __restrict__ emb) {
  __shared__ float st[Dc];
  const int b = blockIdx.y;
  const int tid = threadIdx.x;
  for (int i = tid; i < Dc; i += 256) {
    const float v = t[b * Dc + i];
    st[i] = v / (1.f + __expf(-v));
  }
  __syncthreads();
  const int j = blockIdx.x * 256 + tid;
  float acc = bias[j];
  for (int d = 0; d < Dc; ++d) acc += st[d] * W[(size_t)d * SIXD + j];
  emb[b * SIXD + j] = acc;
}

// ---------------------------------------------------------------------------
// LayerNorm (no affine) + AdaLN modulate -> bf16
// ---------------------------------------------------------------------------
__global__ __launch_bounds__(256) void k_ln_mod(const float* __restrict__ X,
                                                const float* __restrict__ emb,
                                                int shiftOff, int scaleOff,
                                                bf16* __restrict__ out) {
  const int tkn = blockIdx.x;
  const int b = tkn / Nc;
  __shared__ float r1[256], r2[256];
  const int tid = threadIdx.x;
  float s = 0.f, s2 = 0.f;
  for (int i = tid; i < Dc; i += 256) {
    const float v = X[(size_t)tkn * Dc + i];
    s += v; s2 += v * v;
  }
  r1[tid] = s; r2[tid] = s2;
  __syncthreads();
  for (int off = 128; off > 0; off >>= 1) {
    if (tid < off) { r1[tid] += r1[tid + off]; r2[tid] += r2[tid + off]; }
    __syncthreads();
  }
  const float mean = r1[0] * (1.f / Dc);
  const float var  = r2[0] * (1.f / Dc) - mean * mean;
  const float rstd = rsqrtf(var + 1e-6f);
  for (int i = tid; i < Dc; i += 256) {
    const float v  = X[(size_t)tkn * Dc + i];
    const float sc = 1.f + emb[b * SIXD + scaleOff + i];
    const float sh = emb[b * SIXD + shiftOff + i];
    out[(size_t)tkn * Dc + i] = (bf16)((v - mean) * rstd * sc + sh);
  }
}

// ---------------------------------------------------------------------------
// Generic bf16 WMMA GEMM: C[M,N] = A[M,K] * B + bias
//   TRANSB=false: B stored [K,N];  TRANSB=true: B stored [N,K]
//   M,N multiples of 64; K multiple of 32 (guaranteed here).
// Block = 128 threads (4 waves); tile 64x64x32; each wave owns a 32x32 area.
// LDS: A row-major (ld=40), B n-major (ld=40) -> all frag reads ds_load_b128.
// ---------------------------------------------------------------------------
template <bool TRANSB, bool OUTBF>
__global__ __launch_bounds__(128) void k_gemm(const bf16* __restrict__ A,
                                              const bf16* __restrict__ B,
                                              const float* __restrict__ bias,
                                              void* __restrict__ Cout,
                                              int M, int N, int K) {
  __shared__ bf16 As[64 * LDA];
  __shared__ bf16 Bs[64 * LDB];
  const int tid = threadIdx.x, lane = tid & 31, wave = tid >> 5;
  const int wr = (wave >> 1) * 32, wc = (wave & 1) * 32;
  const int tileM = blockIdx.y * 64, tileN = blockIdx.x * 64;
  v8f acc00 = zero8(), acc01 = zero8(), acc10 = zero8(), acc11 = zero8();

  for (int k0 = 0; k0 < K; k0 += 32) {
#pragma unroll
    for (int c = 0; c < 2; ++c) {
      const int chunk = tid + c * 128;
      const int r = chunk >> 2, kk = (chunk & 3) * 8;
      U128 U;
      U.q = *reinterpret_cast<const uint4*>(A + (size_t)(tileM + r) * K + k0 + kk);
      *(v8bf*)(As + r * LDA + kk) = U.v;
    }
#pragma unroll
    for (int c = 0; c < 2; ++c) {
      const int chunk = tid + c * 128;
      if (TRANSB) {
        const int n = chunk >> 2, kk = (chunk & 3) * 8;
        U128 U;
        U.q = *reinterpret_cast<const uint4*>(B + (size_t)(tileN + n) * K + k0 + kk);
        *(v8bf*)(Bs + n * LDB + kk) = U.v;
      } else {
        const int r = chunk >> 3, nn = (chunk & 7) * 8;
        U128 U;
        U.q = *reinterpret_cast<const uint4*>(B + (size_t)(k0 + r) * N + tileN + nn);
#pragma unroll
        for (int j = 0; j < 8; ++j) Bs[(nn + j) * LDB + r] = U.h[j];
      }
    }
    if (k0 + 32 < K) {  // prefetch next k-step (global_prefetch_b8)
      __builtin_prefetch(A + (size_t)(tileM + (tid & 63)) * K + k0 + 32, 0, 1);
      if (TRANSB)
        __builtin_prefetch(B + (size_t)(tileN + (tid & 63)) * K + k0 + 32, 0, 1);
      else
        __builtin_prefetch(B + (size_t)(k0 + 32 + (tid & 31)) * N + tileN, 0, 1);
    }
    __syncthreads();
    const v16bf a0 = fragA(As, LDA, wr + 0, 0, lane);
    const v16bf a1 = fragA(As, LDA, wr + 16, 0, lane);
    const v16bf b0 = fragB(Bs, LDB, wc + 0, 0, lane);
    const v16bf b1 = fragB(Bs, LDB, wc + 16, 0, lane);
    acc00 = wmma_bf16(a0, b0, acc00);
    acc01 = wmma_bf16(a0, b1, acc01);
    acc10 = wmma_bf16(a1, b0, acc10);
    acc11 = wmma_bf16(a1, b1, acc11);
    __syncthreads();
  }

  const int lg = lane >> 4, n = lane & 15;
  float* Cf = (float*)Cout;
  bf16*  Cb = (bf16*)Cout;
#pragma unroll
  for (int i = 0; i < 2; ++i)
#pragma unroll
    for (int j = 0; j < 2; ++j) {
      const v8f acc = (i == 0) ? (j == 0 ? acc00 : acc01)
                               : (j == 0 ? acc10 : acc11);
      const int col = tileN + wc + j * 16 + n;
      const float bv = bias ? bias[col] : 0.f;
#pragma unroll
      for (int r = 0; r < 8; ++r) {
        const int row = tileM + wr + i * 16 + r + 8 * lg;
        const float v = acc[r] + bv;
        if (OUTBF) Cb[(size_t)row * N + col] = (bf16)v;
        else       Cf[(size_t)row * N + col] = v;
      }
    }
}

// ---------------------------------------------------------------------------
// Flash attention: one block = (b, h, 64-query tile); 4 waves; 16 rows/wave.
// QK^T and PV via WMMA; online softmax; key-padding mask.
// ---------------------------------------------------------------------------
__global__ __launch_bounds__(128) void k_attn(const bf16* __restrict__ Qb,
                                              const bf16* __restrict__ Kb,
                                              const bf16* __restrict__ Vb,
                                              const unsigned char* __restrict__ mask,
                                              bf16* __restrict__ Ob) {
  __shared__ bf16 Qs[64 * LDT];   // [query][d]
  __shared__ bf16 Ks[64 * LDT];   // [key][d]   (n-major for S = Q K^T)
  __shared__ bf16 Vt[64 * LDT];   // [d][key]   (n-major for O += P V)
  __shared__ bf16 Ps[64 * LDT];   // [query][key]
  __shared__ float Sb[64 * LDS_S];
  __shared__ float mrow[64], lrow[64], frow[64];
  __shared__ unsigned char mk[64];
  const int tid = threadIdx.x, lane = tid & 31, wave = tid >> 5;
  const int b = blockIdx.z, h = blockIdx.y, q0 = blockIdx.x * 64;
  const int lg = lane >> 4, nn = lane & 15;

#pragma unroll
  for (int c = 0; c < 4; ++c) {           // load 64x64 Q tile (b128 stores)
    const int chunk = tid + c * 128;
    const int r = chunk >> 3, kk = (chunk & 7) * 8;
    U128 U;
    U.q = *reinterpret_cast<const uint4*>(
        Qb + (size_t)(b * Nc + q0 + r) * (Hc * HDc) + h * HDc + kk);
    *(v8bf*)(Qs + r * LDT + kk) = U.v;
  }
  if (tid < 64) { mrow[tid] = -1e30f; lrow[tid] = 0.f; }

  v8f o0 = zero8(), o1 = zero8(), o2 = zero8(), o3 = zero8();

  for (int kt = 0; kt < Nc / 64; ++kt) {
    const int kb = kt * 64;
    __syncthreads();
#pragma unroll
    for (int c = 0; c < 4; ++c) {         // load K (row-major) and V (transposed)
      const int chunk = tid + c * 128;
      const int r = chunk >> 3, kk = (chunk & 7) * 8;
      const size_t off = (size_t)(b * Nc + kb + r) * (Hc * HDc) + h * HDc + kk;
      U128 Uk, Uv;
      Uk.q = *reinterpret_cast<const uint4*>(Kb + off);
      Uv.q = *reinterpret_cast<const uint4*>(Vb + off);
      *(v8bf*)(Ks + r * LDT + kk) = Uk.v;
#pragma unroll
      for (int j = 0; j < 8; ++j) Vt[(kk + j) * LDT + r] = Uv.h[j];
    }
    if (tid < 64) mk[tid] = mask[b * Nc + kb + tid];
    if (kb + 64 < Nc) {
      __builtin_prefetch(
          Kb + (size_t)(b * Nc + kb + 64 + (tid & 63)) * (Hc * HDc) + h * HDc, 0, 1);
      __builtin_prefetch(
          Vb + (size_t)(b * Nc + kb + 64 + (tid & 63)) * (Hc * HDc) + h * HDc, 0, 1);
    }
    __syncthreads();

    // S = (Q K^T) / sqrt(HD) with key mask, to LDS
    const v16bf qa0 = fragA(Qs, LDT, wave * 16, 0, lane);
    const v16bf qa1 = fragA(Qs, LDT, wave * 16, 32, lane);
#pragma unroll
    for (int j = 0; j < 4; ++j) {
      v8f s = zero8();
      s = wmma_bf16(qa0, fragB(Ks, LDT, j * 16, 0, lane), s);
      s = wmma_bf16(qa1, fragB(Ks, LDT, j * 16, 32, lane), s);
#pragma unroll
      for (int r = 0; r < 8; ++r) {
        const int row = wave * 16 + r + 8 * lg;
        const int col = j * 16 + nn;
        float v = s[r] * 0.125f;
        if (!mk[col]) v = -1e30f;
        Sb[row * LDS_S + col] = v;
      }
    }
    __syncthreads();

    // online softmax: lanes 0..15 of each wave own its 16 rows
    if (lane < 16) {
      const int row = wave * 16 + lane;
      const float mold = mrow[row];
      float mx = mold;
#pragma unroll
      for (int c4 = 0; c4 < 16; ++c4) {
        const float4 s4 = *(const float4*)(Sb + row * LDS_S + c4 * 4);
        mx = fmaxf(mx, fmaxf(fmaxf(s4.x, s4.y), fmaxf(s4.z, s4.w)));
      }
      float fac = __expf(mold - mx);
      if (mx <= -1e29f) fac = 1.f;        // everything masked so far
      float ls = 0.f;
#pragma unroll
      for (int c8 = 0; c8 < 8; ++c8) {
        const float4 sa = *(const float4*)(Sb + row * LDS_S + c8 * 8);
        const float4 sb = *(const float4*)(Sb + row * LDS_S + c8 * 8 + 4);
        const float sv[8] = {sa.x, sa.y, sa.z, sa.w, sb.x, sb.y, sb.z, sb.w};
        v8bf pk;
#pragma unroll
        for (int j = 0; j < 8; ++j) {
          const float p = (sv[j] <= -1e29f) ? 0.f : __expf(sv[j] - mx);
          ls += p;
          pk[j] = (bf16)p;
        }
        *(v8bf*)(Ps + row * LDT + c8 * 8) = pk;
      }
      lrow[row] = lrow[row] * fac + ls;
      mrow[row] = mx;
      frow[row] = fac;
    }
    __syncthreads();

    float fr[8];
#pragma unroll
    for (int r = 0; r < 8; ++r) fr[r] = frow[wave * 16 + r + 8 * lg];
#pragma unroll
    for (int r = 0; r < 8; ++r) {
      o0[r] *= fr[r]; o1[r] *= fr[r]; o2[r] *= fr[r]; o3[r] *= fr[r];
    }
    const v16bf pa0 = fragA(Ps, LDT, wave * 16, 0, lane);
    const v16bf pa1 = fragA(Ps, LDT, wave * 16, 32, lane);
    o0 = wmma_bf16(pa0, fragB(Vt, LDT, 0,  0, lane), o0);
    o0 = wmma_bf16(pa1, fragB(Vt, LDT, 0,  32, lane), o0);
    o1 = wmma_bf16(pa0, fragB(Vt, LDT, 16, 0, lane), o1);
    o1 = wmma_bf16(pa1, fragB(Vt, LDT, 16, 32, lane), o1);
    o2 = wmma_bf16(pa0, fragB(Vt, LDT, 32, 0, lane), o2);
    o2 = wmma_bf16(pa1, fragB(Vt, LDT, 32, 32, lane), o2);
    o3 = wmma_bf16(pa0, fragB(Vt, LDT, 48, 0, lane), o3);
    o3 = wmma_bf16(pa1, fragB(Vt, LDT, 48, 32, lane), o3);
  }
  __syncthreads();

  float inv[8];
#pragma unroll
  for (int r = 0; r < 8; ++r) {
    const float l = lrow[wave * 16 + r + 8 * lg];
    inv[r] = (l > 0.f) ? 1.f / l : 0.f;
  }
#pragma unroll
  for (int j = 0; j < 4; ++j) {
    const v8f o = (j == 0) ? o0 : (j == 1) ? o1 : (j == 2) ? o2 : o3;
#pragma unroll
    for (int r = 0; r < 8; ++r) {
      const int row = wave * 16 + r + 8 * lg;
      const int col = j * 16 + nn;
      Ob[(size_t)(b * Nc + q0 + row) * (Hc * HDc) + h * HDc + col] =
          (bf16)(o[r] * inv[r]);
    }
  }
}

// ---------------------------------------------------------------------------
// Residual after attention: x1 = x + gate_msa * (mask ? o_proj : 0)
// ---------------------------------------------------------------------------
__global__ void k_res1(const float* __restrict__ X, const float* __restrict__ oproj,
                       const float* __restrict__ emb,
                       const unsigned char* __restrict__ mask,
                       float* __restrict__ x1) {
  const int idx = blockIdx.x * 256 + threadIdx.x;
  const int tkn = idx >> 10, d = idx & 1023, b = tkn >> 10;
  const float o = mask[tkn] ? oproj[idx] : 0.f;
  x1[idx] = X[idx] + emb[b * SIXD + 2 * Dc + d] * o;
}

// ---------------------------------------------------------------------------
// MoE gating: logits, softmax over 8, top-2 (stable lowest-index ties), count.
// ---------------------------------------------------------------------------
__global__ __launch_bounds__(256) void k_gate(const bf16* __restrict__ xm,
                                              const float* __restrict__ gw,
                                              int* __restrict__ tope,
                                              float* __restrict__ topw,
                                              int* __restrict__ cnt) {
  const int t = blockIdx.x * blockDim.x + threadIdx.x;
  if (t >= Tc) return;
  float acc[Ec];
#pragma unroll
  for (int e = 0; e < Ec; ++e) acc[e] = 0.f;
  for (int d = 0; d < Dc; ++d) {
    const float xv = (float)xm[(size_t)t * Dc + d];
#pragma unroll
    for (int e = 0; e < Ec; ++e) acc[e] += xv * gw[e * Dc + d];
  }
  float mx = acc[0];
#pragma unroll
  for (int e = 1; e < Ec; ++e) mx = fmaxf(mx, acc[e]);
  float sum = 0.f;
  float g[Ec];
#pragma unroll
  for (int e = 0; e < Ec; ++e) { g[e] = __expf(acc[e] - mx); sum += g[e]; }
  const float inv = 1.f / sum;
#pragma unroll
  for (int e = 0; e < Ec; ++e) g[e] *= inv;
  int e1 = 0; float v1 = g[0];
#pragma unroll
  for (int e = 1; e < Ec; ++e) if (g[e] > v1) { v1 = g[e]; e1 = e; }
  int e2 = (e1 == 0) ? 1 : 0; float v2 = g[e2];
#pragma unroll
  for (int e = 0; e < Ec; ++e)
    if (e != e1 && e != e2 && g[e] > v2) { v2 = g[e]; e2 = e; }
  tope[t * 2 + 0] = e1; topw[t * 2 + 0] = v1;
  tope[t * 2 + 1] = e2; topw[t * 2 + 1] = v2;
  atomicAdd(&cnt[e1], 1);
  atomicAdd(&cnt[e2], 1);
}

__global__ void k_scan(const int* __restrict__ cnt, int* __restrict__ basep,
                       int* __restrict__ pcnt) {
  if (threadIdx.x == 0 && blockIdx.x == 0) {
    int tot = 0;
    for (int e = 0; e < Ec; ++e) {
      basep[e] = tot;
      const int pc = (cnt[e] + 63) & ~63;
      pcnt[e] = pc;
      tot += pc;
    }
  }
}

__global__ void k_scatter(const int* __restrict__ tope, const float* __restrict__ topw,
                          const int* __restrict__ basep, int* __restrict__ cursor,
                          int* __restrict__ tok, float* __restrict__ wsl,
                          int* __restrict__ smap) {
  const int i = blockIdx.x * blockDim.x + threadIdx.x;
  if (i >= Tc * Kt) return;
  const int e = tope[i];
  const int p = atomicAdd(&cursor[e], 1);
  const int slot = basep[e] + p;
  tok[slot] = i >> 1;
  wsl[slot] = topw[i];
  smap[i] = slot;
}

__global__ void k_pad(const int* __restrict__ cnt, const int* __restrict__ pcnt,
                      const int* __restrict__ basep, int* __restrict__ tok,
                      float* __restrict__ wsl) {
  const int e = blockIdx.x;
  for (int j = cnt[e] + threadIdx.x; j < pcnt[e]; j += blockDim.x) {
    tok[basep[e] + j] = Tc;   // dummy zero row
    wsl[basep[e] + j] = 0.f;
  }
}

// ---------------------------------------------------------------------------
// MoE stage 1: gathered rows, fused gate+up GEMMs + silu(g)*u*w epilogue.
// Expert weight tiles (Bg/Bu) double-buffered through the TDM: issue tile
// k+1, s_wait_tensorcnt 2 (tile k complete, k+1 in flight), compute tile k.
// ---------------------------------------------------------------------------
__global__ __launch_bounds__(128) void k_moe_gate_up(
    const bf16* __restrict__ X, const bf16* __restrict__ Wg,
    const bf16* __restrict__ Wu, const int* __restrict__ tok,
    const float* __restrict__ wsl, const int* __restrict__ basep,
    const int* __restrict__ pcnt, bf16* __restrict__ Hout) {
  const int e = blockIdx.z;
  const int mt = blockIdx.y;
  if (mt * 64 >= pcnt[e]) return;
  const int tileN = blockIdx.x * 64;
  const int slot0 = basep[e] + mt * 64;
  const bf16* Bg = Wg + (size_t)e * Fc * Dc;
  const bf16* Bu = Wu + (size_t)e * Fc * Dc;

  __shared__ bf16 As[64 * LDA];
  __shared__ bf16 Bgs[2][64 * LDB];
  __shared__ bf16 Bus[2][64 * LDB];
  __shared__ int rowTok[64];
  const int tid = threadIdx.x, lane = tid & 31, wave = tid >> 5;
  const int wr = (wave >> 1) * 32, wc = (wave & 1) * 32;
  if (tid < 64) rowTok[tid] = tok[slot0 + tid];
#if HAVE_TDM
  if (wave == 0) {                        // prologue: prefetch first B tiles
    tdm_load_2d(Bg + (size_t)tileN * Dc, Bgs[0], 64, Dc);
    tdm_load_2d(Bu + (size_t)tileN * Dc, Bus[0], 64, Dc);
  }
#endif
  __syncthreads();

  v8f g00 = zero8(), g01 = zero8(), g10 = zero8(), g11 = zero8();
  v8f u00 = zero8(), u01 = zero8(), u10 = zero8(), u11 = zero8();

#pragma unroll 2
  for (int k0 = 0; k0 < Dc; k0 += 32) {
    const int cur = (k0 >> 5) & 1;
#pragma unroll
    for (int c = 0; c < 2; ++c) {         // gathered A rows (token indices)
      const int chunk = tid + c * 128;
      const int r = chunk >> 2, kk = (chunk & 3) * 8;
      U128 U;
      U.q = *reinterpret_cast<const uint4*>(X + (size_t)rowTok[r] * Dc + k0 + kk);
      *(v8bf*)(As + r * LDA + kk) = U.v;
    }
#if HAVE_TDM
    if (wave == 0) {                      // prefetch next tile, wait current
      if (k0 + 32 < Dc) {
        tdm_load_2d(Bg + (size_t)tileN * Dc + k0 + 32, Bgs[cur ^ 1], 64, Dc);
        tdm_load_2d(Bu + (size_t)tileN * Dc + k0 + 32, Bus[cur ^ 1], 64, Dc);
        tdm_wait2();
      } else {
        tdm_wait0();
      }
    }
#else
#pragma unroll
    for (int c = 0; c < 2; ++c) {         // Bg, Bu ([F,D] n-major)
      const int chunk = tid + c * 128;
      const int n = chunk >> 2, kk = (chunk & 3) * 8;
      U128 Ug, Uu;
      Ug.q = *reinterpret_cast<const uint4*>(Bg + (size_t)(tileN + n) * Dc + k0 + kk);
      Uu.q = *reinterpret_cast<const uint4*>(Bu + (size_t)(tileN + n) * Dc + k0 + kk);
      *(v8bf*)(Bgs[cur] + n * LDB + kk) = Ug.v;
      *(v8bf*)(Bus[cur] + n * LDB + kk) = Uu.v;
    }
    if (k0 + 32 < Dc) {
      __builtin_prefetch(Bg + (size_t)(tileN + (tid & 63)) * Dc + k0 + 32, 0, 1);
      __builtin_prefetch(Bu + (size_t)(tileN + (tid & 63)) * Dc + k0 + 32, 0, 1);
    }
#endif
    __syncthreads();
    const v16bf a0 = fragA(As, LDA, wr + 0, 0, lane);
    const v16bf a1 = fragA(As, LDA, wr + 16, 0, lane);
    const v16bf bg0 = fragB(Bgs[cur], LDB, wc + 0, 0, lane);
    const v16bf bg1 = fragB(Bgs[cur], LDB, wc + 16, 0, lane);
    const v16bf bu0 = fragB(Bus[cur], LDB, wc + 0, 0, lane);
    const v16bf bu1 = fragB(Bus[cur], LDB, wc + 16, 0, lane);
    g00 = wmma_bf16(a0, bg0, g00);  g01 = wmma_bf16(a0, bg1, g01);
    g10 = wmma_bf16(a1, bg0, g10);  g11 = wmma_bf16(a1, bg1, g11);
    u00 = wmma_bf16(a0, bu0, u00);  u01 = wmma_bf16(a0, bu1, u01);
    u10 = wmma_bf16(a1, bu0, u10);  u11 = wmma_bf16(a1, bu1, u11);
    __syncthreads();
  }

  const int lg = lane >> 4, n = lane & 15;
#pragma unroll
  for (int i = 0; i < 2; ++i)
#pragma unroll
    for (int j = 0; j < 2; ++j) {
      const v8f gg = (i == 0) ? (j == 0 ? g00 : g01) : (j == 0 ? g10 : g11);
      const v8f uu = (i == 0) ? (j == 0 ? u00 : u01) : (j == 0 ? u10 : u11);
      const int col = tileN + wc + j * 16 + n;
#pragma unroll
      for (int r = 0; r < 8; ++r) {
        const int row = wr + i * 16 + r + 8 * lg;
        const int slot = slot0 + row;
        const float g = gg[r];
        const float s = g / (1.f + __expf(-g));       // silu
        Hout[(size_t)slot * Fc + col] = (bf16)(s * uu[r] * wsl[slot]);
      }
    }
}

// ---------------------------------------------------------------------------
// MoE stage 2: Y[slot, D] = H[slot, F] x Wd_e^T  (Wd stored [E, D, F], n-major)
// Both A (H rows) and B (Wd) tiles double-buffered through the TDM.
// ---------------------------------------------------------------------------
__global__ __launch_bounds__(128) void k_moe_down(const bf16* __restrict__ Hin,
                                                  const bf16* __restrict__ Wd,
                                                  const int* __restrict__ basep,
                                                  const int* __restrict__ pcnt,
                                                  float* __restrict__ Y) {
  const int e = blockIdx.z;
  const int mt = blockIdx.y;
  if (mt * 64 >= pcnt[e]) return;
  const int tileN = blockIdx.x * 64;
  const int slot0 = basep[e] + mt * 64;
  const bf16* Bm = Wd + (size_t)e * Dc * Fc;

  __shared__ bf16 As[2][64 * LDA];
  __shared__ bf16 Bs[2][64 * LDB];
  const int tid = threadIdx.x, lane = tid & 31, wave = tid >> 5;
  const int wr = (wave >> 1) * 32, wc = (wave & 1) * 32;
  v8f acc00 = zero8(), acc01 = zero8(), acc10 = zero8(), acc11 = zero8();

#if HAVE_TDM
  if (wave == 0) {                        // prologue: prefetch first tiles
    tdm_load_2d(Hin + (size_t)slot0 * Fc, As[0], 64, Fc);
    tdm_load_2d(Bm + (size_t)tileN * Fc, Bs[0], 64, Fc);
  }
#endif

#pragma unroll 2
  for (int k0 = 0; k0 < Fc; k0 += 32) {
    const int cur = (k0 >> 5) & 1;
#if HAVE_TDM
    if (wave == 0) {                      // prefetch next tile, wait current
      if (k0 + 32 < Fc) {
        tdm_load_2d(Hin + (size_t)slot0 * Fc + k0 + 32, As[cur ^ 1], 64, Fc);
        tdm_load_2d(Bm + (size_t)tileN * Fc + k0 + 32, Bs[cur ^ 1], 64, Fc);
        tdm_wait2();
      } else {
        tdm_wait0();
      }
    }
#else
#pragma unroll
    for (int c = 0; c < 2; ++c) {
      const int chunk = tid + c * 128;
      const int r = chunk >> 2, kk = (chunk & 3) * 8;
      U128 U;
      U.q = *reinterpret_cast<const uint4*>(Hin + (size_t)(slot0 + r) * Fc + k0 + kk);
      *(v8bf*)(As[cur] + r * LDA + kk) = U.v;
    }
#pragma unroll
    for (int c = 0; c < 2; ++c) {
      const int chunk = tid + c * 128;
      const int n = chunk >> 2, kk = (chunk & 3) * 8;
      U128 U;
      U.q = *reinterpret_cast<const uint4*>(Bm + (size_t)(tileN + n) * Fc + k0 + kk);
      *(v8bf*)(Bs[cur] + n * LDB + kk) = U.v;
    }
    if (k0 + 32 < Fc) {
      __builtin_prefetch(Hin + (size_t)(slot0 + (tid & 63)) * Fc + k0 + 32, 0, 1);
      __builtin_prefetch(Bm + (size_t)(tileN + (tid & 63)) * Fc + k0 + 32, 0, 1);
    }
#endif
    __syncthreads();
    const v16bf a0 = fragA(As[cur], LDA, wr + 0, 0, lane);
    const v16bf a1 = fragA(As[cur], LDA, wr + 16, 0, lane);
    const v16bf b0 = fragB(Bs[cur], LDB, wc + 0, 0, lane);
    const v16bf b1 = fragB(Bs[cur], LDB, wc + 16, 0, lane);
    acc00 = wmma_bf16(a0, b0, acc00);
    acc01 = wmma_bf16(a0, b1, acc01);
    acc10 = wmma_bf16(a1, b0, acc10);
    acc11 = wmma_bf16(a1, b1, acc11);
    __syncthreads();
  }

  const int lg = lane >> 4, n = lane & 15;
#pragma unroll
  for (int i = 0; i < 2; ++i)
#pragma unroll
    for (int j = 0; j < 2; ++j) {
      const v8f acc = (i == 0) ? (j == 0 ? acc00 : acc01)
                               : (j == 0 ? acc10 : acc11);
      const int col = tileN + wc + j * 16 + n;
#pragma unroll
      for (int r = 0; r < 8; ++r) {
        const int row = wr + i * 16 + r + 8 * lg;
        Y[(size_t)(slot0 + row) * Dc + col] = acc[r];
      }
    }
}

// ---------------------------------------------------------------------------
// Final: out = x1 + gate_mlp * (Y[slot0] + Y[slot1])
// ---------------------------------------------------------------------------
__global__ void k_final(const float* __restrict__ x1, const float* __restrict__ emb,
                        const float* __restrict__ Y, const int* __restrict__ smap,
                        float* __restrict__ out) {
  const int idx = blockIdx.x * 256 + threadIdx.x;
  const int tkn = idx >> 10, d = idx & 1023, b = tkn >> 10;
  const int s0 = smap[tkn * 2 + 0], s1 = smap[tkn * 2 + 1];
  const float y = Y[(size_t)s0 * Dc + d] + Y[(size_t)s1 * Dc + d];
  out[idx] = x1[idx] + emb[b * SIXD + 5 * Dc + d] * y;
}

// ---------------------------------------------------------------------------
// Host orchestration
// ---------------------------------------------------------------------------
extern "C" void kernel_launch(void* const* d_in, const int* in_sizes, int n_in,
                              void* d_out, int out_size, void* d_ws, size_t ws_size,
                              hipStream_t stream) {
  (void)in_sizes; (void)n_in; (void)out_size; (void)ws_size;
  const float* x     = (const float*)d_in[0];
  const float* t     = (const float*)d_in[1];
  const unsigned char* mask = (const unsigned char*)d_in[2];
  const float* ada_w = (const float*)d_in[3];
  const float* ada_b = (const float*)d_in[4];
  const float* wq    = (const float*)d_in[5];
  const float* bq    = (const float*)d_in[6];
  const float* wk    = (const float*)d_in[7];
  const float* bk    = (const float*)d_in[8];
  const float* wv    = (const float*)d_in[9];
  const float* bv    = (const float*)d_in[10];
  const float* wo    = (const float*)d_in[11];
  const float* bo    = (const float*)d_in[12];
  const float* gate_w = (const float*)d_in[13];
  const float* w_gate = (const float*)d_in[14];
  const float* w_up   = (const float*)d_in[15];
  const float* w_down = (const float*)d_in[16];
  float* out = (float*)d_out;

  char* base = (char*)d_ws;
  size_t used = 0;
  auto alloc = [&](size_t bytes) -> void* {
    void* r = base + used;
    used = (used + bytes + 255) & ~(size_t)255;
    return r;
  };

  bf16* wq_b = (bf16*)alloc((size_t)Dc * Dc * 2);
  bf16* wk_b = (bf16*)alloc((size_t)Dc * Dc * 2);
  bf16* wv_b = (bf16*)alloc((size_t)Dc * Dc * 2);
  bf16* wo_b = (bf16*)alloc((size_t)Dc * Dc * 2);
  bf16* wg_b = (bf16*)alloc((size_t)Ec * Fc * Dc * 2);
  bf16* wu_b = (bf16*)alloc((size_t)Ec * Fc * Dc * 2);
  bf16* wd_b = (bf16*)alloc((size_t)Ec * Dc * Fc * 2);
  float* emb = (float*)alloc((size_t)Bc * SIXD * 4);
  bf16* xn   = (bf16*)alloc((size_t)Tc * Dc * 2);
  bf16* qb   = (bf16*)alloc((size_t)Tc * Dc * 2);
  bf16* kb   = (bf16*)alloc((size_t)Tc * Dc * 2);
  bf16* vb   = (bf16*)alloc((size_t)Tc * Dc * 2);
  bf16* ob   = (bf16*)alloc((size_t)Tc * Dc * 2);
  float* oproj = (float*)alloc((size_t)Tc * Dc * 4);
  float* x1    = (float*)alloc((size_t)Tc * Dc * 4);
  bf16* xm     = (bf16*)alloc((size_t)(Tc + 1) * Dc * 2);   // +1 dummy zero row
  int*   tope  = (int*)alloc((size_t)Tc * Kt * 4);
  float* topw  = (float*)alloc((size_t)Tc * Kt * 4);
  int*   meta  = (int*)alloc(32 * 4);  // cnt[8] cursor[8] base[8] pcnt[8]
  int*   cnt = meta, *cursor = meta + 8, *basep = meta + 16, *pcnt = meta + 24;
  int*   tok  = (int*)alloc((size_t)MAXSLOTS * 4);
  float* wsl  = (float*)alloc((size_t)MAXSLOTS * 4);
  int*   smap = (int*)alloc((size_t)Tc * Kt * 4);
  bf16*  Hbuf = (bf16*)alloc((size_t)MAXSLOTS * Fc * 2);
  float* Ybuf = (float*)alloc((size_t)MAXSLOTS * Dc * 4);

  // ---- weight conversion to bf16 (one pass, bandwidth trivial) ----
  k_cvt<<<1024, 256, 0, stream>>>(wq, wq_b, (size_t)Dc * Dc);
  k_cvt<<<1024, 256, 0, stream>>>(wk, wk_b, (size_t)Dc * Dc);
  k_cvt<<<1024, 256, 0, stream>>>(wv, wv_b, (size_t)Dc * Dc);
  k_cvt<<<1024, 256, 0, stream>>>(wo, wo_b, (size_t)Dc * Dc);
  k_cvt<<<2048, 256, 0, stream>>>(w_gate, wg_b, (size_t)Ec * Fc * Dc);
  k_cvt<<<2048, 256, 0, stream>>>(w_up,   wu_b, (size_t)Ec * Fc * Dc);
  k_cvt<<<2048, 256, 0, stream>>>(w_down, wd_b, (size_t)Ec * Dc * Fc);

  // ---- AdaLN ----
  k_adaln<<<dim3(SIXD / 256, Bc), 256, 0, stream>>>(t, ada_w, ada_b, emb);
  k_ln_mod<<<Tc, 256, 0, stream>>>(x, emb, /*shift*/0, /*scale*/Dc, xn);

  // ---- QKV projections ----
  const dim3 gproj(Dc / 64, Tc / 64);
  k_gemm<false, true><<<gproj, 128, 0, stream>>>(xn, wq_b, bq, qb, Tc, Dc, Dc);
  k_gemm<false, true><<<gproj, 128, 0, stream>>>(xn, wk_b, bk, kb, Tc, Dc, Dc);
  k_gemm<false, true><<<gproj, 128, 0, stream>>>(xn, wv_b, bv, vb, Tc, Dc, Dc);

  // ---- attention ----
  k_attn<<<dim3(Nc / 64, Hc, Bc), 128, 0, stream>>>(qb, kb, vb, mask, ob);

  // ---- output projection + gated residual ----
  k_gemm<false, false><<<gproj, 128, 0, stream>>>(ob, wo_b, bo, oproj, Tc, Dc, Dc);
  k_res1<<<(Tc * Dc) / 256, 256, 0, stream>>>(x, oproj, emb, mask, x1);

  // ---- second AdaLN + MoE routing ----
  k_ln_mod<<<Tc, 256, 0, stream>>>(x1, emb, 3 * Dc, 4 * Dc, xm);
  k_zero_bf16<<<4, 256, 0, stream>>>(xm + (size_t)Tc * Dc, Dc);  // dummy row
  k_zero_i32<<<1, 32, 0, stream>>>(meta, 16);                    // cnt + cursor
  k_gate<<<(Tc + 255) / 256, 256, 0, stream>>>(xm, gate_w, tope, topw, cnt);
  k_scan<<<1, 1, 0, stream>>>(cnt, basep, pcnt);
  k_scatter<<<(Tc * Kt + 255) / 256, 256, 0, stream>>>(tope, topw, basep, cursor,
                                                       tok, wsl, smap);
  k_pad<<<Ec, 64, 0, stream>>>(cnt, pcnt, basep, tok, wsl);

  // ---- MoE expert GEMMs (gathered, top-2 only: 4x fewer FLOPs than dense) ----
  k_moe_gate_up<<<dim3(Fc / 64, Tc / 64, Ec), 128, 0, stream>>>(
      xm, wg_b, wu_b, tok, wsl, basep, pcnt, Hbuf);
  k_moe_down<<<dim3(Dc / 64, Tc / 64, Ec), 128, 0, stream>>>(
      Hbuf, wd_b, basep, pcnt, Ybuf);

  // ---- final gated residual ----
  k_final<<<(Tc * Dc) / 256, 256, 0, stream>>>(x1, emb, Ybuf, smap, out);
}
